// Deepseekv2MoE_70016556860061
// MI455X (gfx1250) — matrix operand order repaired and checked by reading, not compiled, not run on hardware
//
#include <hip/hip_runtime.h>

#define E 32
#define G 8
#define TG 3
#define TOPK 6
#define H 2048
#define II 1408
#define T 1024

typedef __attribute__((ext_vector_type(16))) __bf16 v16bf;
typedef __attribute__((ext_vector_type(8)))  float  v8f;

union FragU {
    v16bf    v;
    unsigned u[8];
    uint4    q[2];
};

// ---------- helpers ----------
// pack two fp32 -> one dword of two bf16 (v_cvt_pk_bf16_f32)
static __device__ __forceinline__ unsigned pk2(float lo, float hi) {
#if __has_builtin(__builtin_amdgcn_cvt_pk_bf16_f32)
    auto r = __builtin_amdgcn_cvt_pk_bf16_f32(lo, hi);
    return __builtin_bit_cast(unsigned, r);
#else
    unsigned out;
    asm("v_cvt_pk_bf16_f32 %0, %1, %2" : "=v"(out) : "v"(lo), "v"(hi));
    return out;
#endif
}
static __device__ __forceinline__ unsigned short f2bfbits(float f) {
    unsigned u = __builtin_bit_cast(unsigned, f);
    unsigned r = u + 0x7FFFu + ((u >> 16) & 1u);   // RNE to bf16
    return (unsigned short)(r >> 16);
}
// B fragment: 16 contiguous fp32 -> v16bf (per-lane K-contiguous weight row)
static __device__ __forceinline__ v16bf ldB(const float* __restrict__ p) {
    float4 q0 = *(const float4*)(p);
    float4 q1 = *(const float4*)(p + 4);
    float4 q2 = *(const float4*)(p + 8);
    float4 q3 = *(const float4*)(p + 12);
    FragU f;
    f.u[0] = pk2(q0.x, q0.y); f.u[1] = pk2(q0.z, q0.w);
    f.u[2] = pk2(q1.x, q1.y); f.u[3] = pk2(q1.z, q1.w);
    f.u[4] = pk2(q2.x, q2.y); f.u[5] = pk2(q2.z, q2.w);
    f.u[6] = pk2(q3.x, q3.y); f.u[7] = pk2(q3.z, q3.w);
    return f.v;
}
// A fragment from fp32: 16 contiguous values [p..p+15] (K-contiguous, matches ldB)
static __device__ __forceinline__ v16bf ldA(const float* __restrict__ p) {
    float4 q0 = *(const float4*)(p);
    float4 q1 = *(const float4*)(p + 4);
    float4 q2 = *(const float4*)(p + 8);
    float4 q3 = *(const float4*)(p + 12);
    FragU f;
    f.u[0] = pk2(q0.x, q0.y); f.u[1] = pk2(q0.z, q0.w);
    f.u[2] = pk2(q1.x, q1.y); f.u[3] = pk2(q1.z, q1.w);
    f.u[4] = pk2(q2.x, q2.y); f.u[5] = pk2(q2.z, q2.w);
    f.u[6] = pk2(q3.x, q3.y); f.u[7] = pk2(q3.z, q3.w);
    return f.v;
}
// A fragment from packed bf16 memory: pure 2x16B load + bitcast (no VALU)
static __device__ __forceinline__ v16bf ldAbf(const unsigned short* __restrict__ p) {
    FragU f;
    f.q[0] = *(const uint4*)(p);        // K [0..7]
    f.q[1] = *(const uint4*)(p + 8);    // K [8..15]
    return f.v;
}
static __device__ __forceinline__ v8f wmma_bf16(v16bf a, v16bf b, v8f c) {
    return __builtin_amdgcn_wmma_f32_16x16x32_bf16(false, a, false, b,
                                                   (short)0, c, false, false);
}
static __device__ __forceinline__ float silu(float x) {
    return x * __builtin_amdgcn_rcpf(1.f + __expf(-x));
}

// ---------- 1) router: one wave32 per token, lane == expert ----------
__global__ __launch_bounds__(32) void moe_router(const float* __restrict__ x,
                                                 const float* __restrict__ gw,
                                                 float* __restrict__ rw) {
    const int t    = blockIdx.x;
    const int lane = threadIdx.x;                 // expert id
    const float4* xv = (const float4*)(x + (size_t)t * H);
    const float4* gv = (const float4*)(gw + (size_t)lane * H);
    float acc = 0.f;
    for (int k = 0; k < H / 4; ++k) {
        float4 a = xv[k], b = gv[k];
        acc += a.x * b.x + a.y * b.y + a.z * b.z + a.w * b.w;
    }
    float m = acc;
    for (int o = 16; o >= 1; o >>= 1) m = fmaxf(m, __shfl_xor(m, o, 32));
    float p = __expf(acc - m);
    float s = p;
    for (int o = 16; o >= 1; o >>= 1) s += __shfl_xor(s, o, 32);
    p /= s;
    float gs = p;
    gs = fmaxf(gs, __shfl_xor(gs, 1, 32));
    gs = fmaxf(gs, __shfl_xor(gs, 2, 32));
    const int gid = lane >> 2;
    int grank = 0;
    for (int g = 0; g < G; ++g) {
        float sg = __shfl(gs, g * 4, 32);
        grank += (sg > gs) || (sg == gs && g < gid);
    }
    float tp = (grank < TG) ? p : 0.f;
    int r = 0;
    for (int j = 0; j < E; ++j) {
        float tj = __shfl(tp, j, 32);
        r += (tj > tp) || (tj == tp && j < lane);
    }
    rw[t * E + lane] = (r < TOPK && tp > 0.f) ? tp : 0.f;
}

// ---------- 2) count assignments per expert ----------
__global__ void moe_count(const float* __restrict__ rw, int* __restrict__ cnt) {
    int i = blockIdx.x * blockDim.x + threadIdx.x;
    if (i < T * E && rw[i] > 0.f) atomicAdd(&cnt[i & (E - 1)], 1);
}

// ---------- 3) exclusive scan of 32 counts (one wave) ----------
__global__ __launch_bounds__(32) void moe_scan(const int* __restrict__ cnt,
                                               int* __restrict__ offs) {
    int lane = threadIdx.x;
    int v = cnt[lane];
    for (int o = 1; o < 32; o <<= 1) {
        int n = __shfl_up(v, o, 32);
        if (lane >= o) v += n;
    }
    if (lane == 0) offs[0] = 0;
    offs[lane + 1] = v;
}

// ---------- 4) compact per-expert (token, weight) lists ----------
__global__ void moe_fill(const float* __restrict__ rw, const int* __restrict__ offs,
                         int* __restrict__ cnt2, int* __restrict__ rows,
                         float* __restrict__ wgt) {
    int i = blockIdx.x * blockDim.x + threadIdx.x;       // i = t*E + e
    if (i >= T * E) return;
    float w = rw[i];
    if (w <= 0.f) return;
    int e = i & (E - 1), t = i >> 5;
    int pos = atomicAdd(&cnt2[e], 1);
    int idx = offs[e] + pos;
    rows[idx] = t;
    wgt[idx] = w;
}

// ---------- 5) grouped GEMM-1: hact = silu(X*W1^T) * (X*W3^T) ----------
// One wave computes a 32(M) x 32(N) tile for both W1 and W3 (8 WMMA / K-step).
__global__ __launch_bounds__(32) void moe_gemm1(const float* __restrict__ x,
                                                const float* __restrict__ w1,
                                                const float* __restrict__ w3,
                                                const int* __restrict__ offs,
                                                const int* __restrict__ rows,
                                                unsigned short* __restrict__ hact) {
    const int e  = blockIdx.z;
    const int tm = blockIdx.y;               // 32-row tile of this expert's tokens
    const int ti = blockIdx.x;               // 32-col tile of I
    const int base = offs[e];
    const int Ne   = offs[e + 1] - base;
    if (tm * 32 >= Ne) return;

    const int lane = threadIdx.x;
    const int half = lane >> 4;
    const int nl   = lane & 15;

    int rA0 = tm * 32 + nl;       if (rA0 >= Ne) rA0 = Ne - 1;
    int rA1 = tm * 32 + 16 + nl;  if (rA1 >= Ne) rA1 = Ne - 1;
    const int tok0 = rows[base + rA0];
    const int tok1 = rows[base + rA1];

    const float* xr0 = x + (size_t)tok0 * H + half * 16;
    const float* xr1 = x + (size_t)tok1 * H + half * 16;
    const int i0 = ti * 32;
    const float* w1r0 = w1 + ((size_t)e * II + (size_t)(i0 + nl)) * H + half * 16;
    const float* w1r1 = w1 + ((size_t)e * II + (size_t)(i0 + 16 + nl)) * H + half * 16;
    const float* w3r0 = w3 + ((size_t)e * II + (size_t)(i0 + nl)) * H + half * 16;
    const float* w3r1 = w3 + ((size_t)e * II + (size_t)(i0 + 16 + nl)) * H + half * 16;

    v8f g00 = {}, g01 = {}, g10 = {}, g11 = {};   // W1 path
    v8f u00 = {}, u01 = {}, u10 = {}, u11 = {};   // W3 path
    for (int kb = 0; kb < H; kb += 32) {
        __builtin_prefetch(w1r0 + kb + 256, 0, 1);
        __builtin_prefetch(w3r0 + kb + 256, 0, 1);
        v16bf a0 = ldA(xr0 + kb);
        v16bf a1 = ldA(xr1 + kb);
        v16bf b0 = ldB(w1r0 + kb);
        v16bf b1 = ldB(w1r1 + kb);
        g00 = wmma_bf16(a0, b0, g00);
        g01 = wmma_bf16(a0, b1, g01);
        g10 = wmma_bf16(a1, b0, g10);
        g11 = wmma_bf16(a1, b1, g11);
        v16bf c0 = ldB(w3r0 + kb);
        v16bf c1 = ldB(w3r1 + kb);
        u00 = wmma_bf16(a0, c0, u00);
        u01 = wmma_bf16(a0, c1, u01);
        u10 = wmma_bf16(a1, c0, u10);
        u11 = wmma_bf16(a1, c1, u11);
    }
#pragma unroll
    for (int mi = 0; mi < 2; ++mi) {
        const v8f* gm = mi ? &g10 : &g00;   // [ni] selected below
        const v8f* um = mi ? &u10 : &u00;
#pragma unroll
        for (int ni = 0; ni < 2; ++ni) {
            v8f gv = ni ? (mi ? g11 : g01) : (mi ? g10 : g00);
            v8f uv = ni ? (mi ? u11 : u01) : (mi ? u10 : u00);
            (void)gm; (void)um;
#pragma unroll
            for (int v = 0; v < 8; ++v) {
                int mr = mi * 16 + half * 8 + v;
                int r  = tm * 32 + mr;
                if (r < Ne) {
                    float hv = silu(gv[v]) * uv[v];
                    hact[(size_t)(base + r) * II + i0 + ni * 16 + nl] = f2bfbits(hv);
                }
            }
        }
    }
}

// ---------- 6) grouped GEMM-2: out += (hact * W2^T) * route_w ----------
// One wave computes a 32(M) x 32(N) tile (4 WMMA / K-step); A is already bf16.
__global__ __launch_bounds__(32) void moe_gemm2(const unsigned short* __restrict__ hact,
                                                const float* __restrict__ w2,
                                                const int* __restrict__ offs,
                                                const int* __restrict__ rows,
                                                const float* __restrict__ wgt,
                                                float* __restrict__ out) {
    const int e  = blockIdx.z;
    const int tm = blockIdx.y;
    const int th = blockIdx.x;               // 32-col tile of H
    const int base = offs[e];
    const int Ne   = offs[e + 1] - base;
    if (tm * 32 >= Ne) return;

    const int lane = threadIdx.x;
    const int half = lane >> 4;
    const int nl   = lane & 15;

    int rA0 = tm * 32 + nl;       if (rA0 >= Ne) rA0 = Ne - 1;
    int rA1 = tm * 32 + 16 + nl;  if (rA1 >= Ne) rA1 = Ne - 1;
    const unsigned short* ar0 = hact + (size_t)(base + rA0) * II + half * 16;
    const unsigned short* ar1 = hact + (size_t)(base + rA1) * II + half * 16;

    const float* w2e = w2 + (size_t)e * H * II;
    const int h0 = th * 32;
    const float* b0r = w2e + (size_t)(h0 + nl) * II + half * 16;
    const float* b1r = w2e + (size_t)(h0 + 16 + nl) * II + half * 16;

    v8f a00 = {}, a01 = {}, a10 = {}, a11 = {};
    for (int kb = 0; kb < II; kb += 32) {
        __builtin_prefetch(b0r + kb + 256, 0, 1);
        __builtin_prefetch(b1r + kb + 256, 0, 1);
        v16bf fa0 = ldAbf(ar0 + kb);
        v16bf fa1 = ldAbf(ar1 + kb);
        v16bf fb0 = ldB(b0r + kb);
        v16bf fb1 = ldB(b1r + kb);
        a00 = wmma_bf16(fa0, fb0, a00);
        a01 = wmma_bf16(fa0, fb1, a01);
        a10 = wmma_bf16(fa1, fb0, a10);
        a11 = wmma_bf16(fa1, fb1, a11);
    }
#pragma unroll
    for (int mi = 0; mi < 2; ++mi) {
#pragma unroll
        for (int v = 0; v < 8; ++v) {
            int mr = mi * 16 + half * 8 + v;
            int r  = tm * 32 + mr;
            if (r < Ne) {
                int   slot = base + r;
                int   t    = rows[slot];
                float w    = wgt[slot];
                v8f c0 = mi ? a10 : a00;
                v8f c1 = mi ? a11 : a01;
                atomicAdd(&out[(size_t)t * H + h0 + nl],      c0[v] * w);
                atomicAdd(&out[(size_t)t * H + h0 + 16 + nl], c1[v] * w);
            }
        }
    }
}

// ---------- host-side launch ----------
extern "C" void kernel_launch(void* const* d_in, const int* in_sizes, int n_in,
                              void* d_out, int out_size, void* d_ws, size_t ws_size,
                              hipStream_t stream) {
    const float* x  = (const float*)d_in[0];   // [T, H]
    const float* gw = (const float*)d_in[1];   // [E, H]
    const float* w1 = (const float*)d_in[2];   // [E, I, H]
    const float* w3 = (const float*)d_in[3];   // [E, I, H]
    const float* w2 = (const float*)d_in[4];   // [E, H, I]
    float* out = (float*)d_out;                // [T, H]

    char* wsb = (char*)d_ws;
    const size_t OFF_RW   = 0;
    const size_t OFF_CNT  = OFF_RW   + (size_t)T * E * 4;
    const size_t OFF_CNT2 = OFF_CNT  + 128;
    const size_t OFF_OFFS = OFF_CNT2 + 128;
    const size_t OFF_ROWS = OFF_OFFS + 256;
    const size_t OFF_WGT  = OFF_ROWS + (size_t)T * TOPK * 4;
    const size_t OFF_HACT = OFF_WGT  + (size_t)T * TOPK * 4;   // T*K*I bf16 (~17.3 MB)

    float*          rw   = (float*)(wsb + OFF_RW);
    int*            cnt  = (int*)  (wsb + OFF_CNT);
    int*            cnt2 = (int*)  (wsb + OFF_CNT2);
    int*            offs = (int*)  (wsb + OFF_OFFS);
    int*            rows = (int*)  (wsb + OFF_ROWS);
    float*          wgt  = (float*)(wsb + OFF_WGT);
    unsigned short* hact = (unsigned short*)(wsb + OFF_HACT);

    hipMemsetAsync(wsb + OFF_CNT, 0, 512, stream);
    hipMemsetAsync(out, 0, (size_t)T * H * sizeof(float), stream);

    moe_router<<<T, 32, 0, stream>>>(x, gw, rw);
    moe_count <<<(T * E + 255) / 256, 256, 0, stream>>>(rw, cnt);
    moe_scan  <<<1, 32, 0, stream>>>(cnt, offs);
    moe_fill  <<<(T * E + 255) / 256, 256, 0, stream>>>(rw, offs, cnt2, rows, wgt);

    moe_gemm1<<<dim3(II / 32, T / 32, E), 32, 0, stream>>>(x, w1, w3, offs, rows, hact);
    moe_gemm2<<<dim3(H / 32, T / 32, E), 32, 0, stream>>>(hact, w2, offs, rows, wgt, out);
}
